// AdaptiveFieldEvolution_55276229100100
// MI455X (gfx1250) — matrix operand order repaired and checked by reading, not compiled, not run on hardware
//
#include <hip/hip_runtime.h>
#include <hip/hip_bf16.h>
#include <cstdint>

// ---------------------------------------------------------------------------
// Adaptive field evolution on MI455X (gfx1250):
//   50 Euler steps of (3x3 SAME conv + tanh), implicit-GEMM on bf16 WMMA.
//   M = B*H*W = 65536, N = 128, K = 9*128 = 1152 (36 chunks of K=32).
//   A-panel staged into LDS via GLOBAL_LOAD_ASYNC_TO_LDS_B128 from a
//   zero-padded bf16 shadow state (no bounds checks, no conversion in the
//   hot path). fp32 master state updated in place for accuracy.
// ---------------------------------------------------------------------------

#define DT_       0.1f
#define THRESH_   0.01f
#define B_        16
#define C_        128
#define H_        64
#define W_        64
#define NCHUNK_   36          // K chunks of 32
#define ELEMS_    (B_*C_*H_*W_)
#define PH_       66          // padded height
#define PW_       66          // padded width
#define SHADOW_ELEMS_ (B_*PH_*PW_*C_)          // bf16 elems per shadow
#define SHADOW_BYTES_ ((size_t)SHADOW_ELEMS_*2)

typedef __bf16 bf16_t;
typedef bf16_t v16bf __attribute__((ext_vector_type(16)));
typedef float  v8f   __attribute__((ext_vector_type(8)));

union ABRegs { uint32_t u[8]; v16bf v; };

__device__ __forceinline__ uint16_t f2bf(float f) {
    uint32_t u = __builtin_bit_cast(uint32_t, f);
    u += 0x7FFFu + ((u >> 16) & 1u);
    return (uint16_t)(u >> 16);
}
__device__ __forceinline__ uint32_t pack_bf16x2(float a, float b) {
    return (uint32_t)f2bf(a) | ((uint32_t)f2bf(b) << 16);
}

// ---------------------------------------------------------------------------
// Zero both bf16 shadow buffers (borders must be zero; interior overwritten).
// ---------------------------------------------------------------------------
__global__ void zero_shadows_kernel(uint32_t* __restrict__ p, int ndwords)
{
    int e = blockIdx.x * 256 + threadIdx.x;
    if (e < ndwords) p[e] = 0u;
}

// ---------------------------------------------------------------------------
// Init: fp32 state (NHWC, in-place buffer) + bf16 padded shadow interior
//       + bookkeeping state.
// ---------------------------------------------------------------------------
__global__ void init_state_kernel(const float* __restrict__ field,
                                  float*    __restrict__ cur,
                                  uint16_t* __restrict__ shadowA,
                                  float*    __restrict__ init_change,
                                  float*    __restrict__ steps,
                                  int*      __restrict__ converged,
                                  float     max_steps_f)
{
    int e = blockIdx.x * 256 + threadIdx.x;          // NHWC linear index
    if (e < ELEMS_) {
        int c = e & 127;
        int w = (e >> 7)  & 63;
        int h = (e >> 13) & 63;
        int b =  e >> 19;
        float v = field[(((b * 128 + c) * 64 + h) * 64) + w];
        cur[e] = v;
        shadowA[(((b * PH_ + (h + 1)) * PW_ + (w + 1)) * 128) + c] = f2bf(v);
    }
    if (blockIdx.x == 0 && threadIdx.x < B_) {
        init_change[threadIdx.x] = 1.0f;
        steps[threadIdx.x]       = max_steps_f;
        converged[threadIdx.x]   = 0;
    }
}

// ---------------------------------------------------------------------------
// Pack W (OIHW fp32) into WMMA B-matrix order, bf16.
// dword d = ((ntile*36 + chunk)*32 + lane)*8 + i
//   lane 0-15  : n = ntile*16 + lane,     k_local = 2*i
//   lane 16-31 : n = ntile*16 + lane-16,  k_local = 16 + 2*i
// ---------------------------------------------------------------------------
__global__ void pack_weights_kernel(const float* __restrict__ Wsrc,
                                    uint32_t* __restrict__ bpack)
{
    int d = blockIdx.x * 256 + threadIdx.x;
    if (d >= 8 * NCHUNK_ * 256) return;
    int i     =  d & 7;
    int lane  = (d >> 3) & 31;
    int chunk = (d >> 8) % NCHUNK_;
    int ntile =  d / (NCHUNK_ * 256);
    int n  = ntile * 16 + (lane & 15);
    int kl = ((lane >> 4) << 4) + (i << 1);
    int k  = chunk * 32 + kl;
    int tap = k >> 7;            // 0..8 = ky*3+kx
    int cin = k & 127;           // even
    float w0 = Wsrc[(n * 128 + cin)     * 9 + tap];
    float w1 = Wsrc[(n * 128 + cin + 1) * 9 + tap];
    bpack[d] = pack_bf16x2(w0, w1);
}

// ---------------------------------------------------------------------------
// One evolution step. Block = 256 threads = 8 waves.
// Block -> (b, h, w0): 16 pixels of one row; wave wv -> channels [16wv,+16).
// A panel: async DMA from padded bf16 shadow directly into WMMA A layout.
// ---------------------------------------------------------------------------
__global__ __launch_bounds__(256)
void step_kernel(float*       __restrict__ cur,           // fp32 state, in-place
                 const uint32_t* __restrict__ shadowSrc,  // padded bf16 (dwords)
                 uint16_t*    __restrict__ shadowDst,     // padded bf16 (u16)
                 const uint32_t* __restrict__ bpack,
                 const float* __restrict__ bias,
                 const int*   __restrict__ converged,
                 float*       __restrict__ partial,
                 float*       __restrict__ finals)
{
    __shared__ uint32_t sA[NCHUNK_ * 256];   // 36 KB: A panel in WMMA order
    __shared__ float    sRed[256];

    const int tid = threadIdx.x;
    const int blk = blockIdx.x;
    const int b   = blk >> 8;                // 256 blocks per image
    const int rem = blk & 255;
    const int h   = rem >> 2;
    const int w0  = (rem & 3) << 4;

    // --- stage A panel via async global->LDS DMA --------------------------
    // Transfer unit: 16B = 4 channel-pairs, contiguous in global AND in the
    // WMMA A layout. 9 x b128 per thread covers 36 chunks x 256 dwords.
    //   q = tid & 63 : half2 = q & 1, lane_s = q >> 1 (A-layout lane)
    //   lane_s = lhalf*16 + m ; chunk = it*4 + (tid>>6)
    {
        const int q      = tid & 63;
        const int half2  = q & 1;
        const int lane_s = q >> 1;
        const int ms     = lane_s & 15;
        const int lhalf  = lane_s >> 4;
        const int csub   = tid >> 6;                      // chunk sub-index 0..3
        const uint32_t ldsBase =
            (uint32_t)(uintptr_t)(&sA[0]);                // LDS aperture: low 32 bits
        #pragma unroll
        for (int it = 0; it < 9; ++it) {
            const int chunk = it * 4 + csub;
            const int tap   = chunk >> 2;                 // uniform per chunk
            const int ky    = tap / 3;
            const int kx    = tap - 3 * ky;
            const int cp0   = ((chunk & 3) << 4) + (lhalf << 3) + (half2 << 2);
            const uint32_t* gsrc = shadowSrc +
                ((size_t)((b * PH_ + (h + ky)) * PW_ + (w0 + ms + kx)) * 64 + cp0);
            const uint32_t ldsaddr =
                ldsBase + (uint32_t)(chunk * 1024 + lane_s * 32 + half2 * 16);
            asm volatile("global_load_async_to_lds_b128 %0, %1, off"
                         :: "v"(ldsaddr), "v"(gsrc) : "memory");
        }
        asm volatile("s_wait_asynccnt 0x0" ::: "memory");
    }
    __syncthreads();

    // --- WMMA main loop ---------------------------------------------------
    const int wv   = tid >> 5;               // channel tile 0..7
    const int lane = tid & 31;
    v8f acc = {};

    const uint32_t* gbase = bpack + ((size_t)(wv * NCHUNK_) * 256) + lane * 8;
    const uint32_t* sbase = sA + lane * 8;

    #pragma unroll 4
    for (int chunk = 0; chunk < NCHUNK_; ++chunk) {
        ABRegs A, Bm;
        const uint32_t* sa = sbase + chunk * 256;
        const uint32_t* gb = gbase + chunk * 256;
        #pragma unroll
        for (int i = 0; i < 8; ++i) A.u[i]  = sa[i];   // -> ds_load_b128 x2
        #pragma unroll
        for (int i = 0; i < 8; ++i) Bm.u[i] = gb[i];   // -> global_load_b128 x2
        acc = __builtin_amdgcn_wmma_f32_16x16x32_bf16(
                  false, A.v, false, Bm.v, (short)0, acc, false, false);
    }

    // --- epilogue: tanh, Euler update, shadow + finals, |delta| sum -------
    // C/D layout: lane l, vgpr r -> M = (l>>4)*8 + r, N = l&15
    const int nloc = lane & 15;
    const int mhi  = (lane >> 4) << 3;
    const int ch   = wv * 16 + nloc;
    const float bv = bias[ch];
    const int convb = converged[b];
    float lsum = 0.0f;

    #pragma unroll
    for (int r = 0; r < 8; ++r) {
        int m = mhi + r;
        int w = w0 + m;
        float y  = acc[r] + bv;
        float th = tanhf(y);
        int nidx = (((b * 64 + h) * 64 + w) * 128) + ch;
        float cv = cur[nidx];
        float nv = cv + DT_ * (th - cv);
        cur[nidx] = nv;                                  // in-place fp32 state
        shadowDst[(((b * PH_ + (h + 1)) * PW_ + (w + 1)) * 128) + ch] = f2bf(nv);
        if (!convb)
            finals[(((b * 128 + ch) * 64 + h) * 64) + w] = nv;
        lsum += fabsf(nv - cv);
    }

    sRed[tid] = lsum;
    __syncthreads();
    for (int s = 128; s > 0; s >>= 1) {
        if (tid < s) sRed[tid] += sRed[tid + s];
        __syncthreads();
    }
    if (tid == 0) partial[blk] = sRed[0];    // deterministic (no atomics)
}

// ---------------------------------------------------------------------------
// Per-step bookkeeping: fold 256 partials/batch, convergence state, history.
// ---------------------------------------------------------------------------
__global__ void book_kernel(const float* __restrict__ partial,
                            float* __restrict__ init_change,
                            float* __restrict__ steps,
                            int*   __restrict__ converged,
                            float* __restrict__ out_hist,
                            int step_idx)
{
    __shared__ float red[256];
    __shared__ float chg[B_];
    int t = threadIdx.x;
    for (int b = 0; b < B_; ++b) {
        red[t] = partial[b * 256 + t];
        __syncthreads();
        for (int s = 128; s > 0; s >>= 1) {
            if (t < s) red[t] += red[t + s];
            __syncthreads();
        }
        if (t == 0) chg[b] = red[0] * (1.0f / (float)(C_ * H_ * W_));
        __syncthreads();
    }
    if (t == 0) {
        float hsum = 0.0f;
        for (int b = 0; b < B_; ++b) {
            float change = chg[b];
            hsum += change;
            if (step_idx == 0) init_change[b] = fmaxf(change, 1e-8f);
            float rel = change / init_change[b];
            if (!converged[b] && rel < THRESH_) {
                steps[b]     = (float)(step_idx + 1);
                converged[b] = 1;
            }
        }
        out_hist[step_idx] = hsum * (1.0f / (float)B_);
    }
}

// ---------------------------------------------------------------------------
// Final: steps vector + ponder cost.
// ---------------------------------------------------------------------------
__global__ void finalize_kernel(const float* __restrict__ steps,
                                float* __restrict__ out_steps,
                                float* __restrict__ out_ponder,
                                float  max_steps_f)
{
    int t = threadIdx.x;
    if (t < B_) out_steps[t] = steps[t];
    if (t == 0) {
        float s = 0.0f;
        for (int b = 0; b < B_; ++b) s += steps[b];
        out_ponder[0] = (s / (float)B_) / max_steps_f;
    }
}

// ---------------------------------------------------------------------------
extern "C" void kernel_launch(void* const* d_in, const int* in_sizes, int n_in,
                              void* d_out, int out_size, void* d_ws, size_t ws_size,
                              hipStream_t stream)
{
    (void)in_sizes; (void)n_in; (void)ws_size;
    const float* field = (const float*)d_in[0];
    const float* Wsrc  = (const float*)d_in[1];
    const float* bias  = (const float*)d_in[2];

    // recover max_steps from output layout: finals + steps + ponder + history
    int max_steps = out_size - (ELEMS_ + B_ + 1);
    if (max_steps < 1) max_steps = 50;
    const float msf = (float)max_steps;

    char* ws = (char*)d_ws;
    size_t off = 0;
    float*    cur     = (float*)(ws + off);        off += (size_t)ELEMS_ * 4;      // 32 MB
    uint16_t* shadowA = (uint16_t*)(ws + off);     off += SHADOW_BYTES_;           // ~17 MB
    uint16_t* shadowB = (uint16_t*)(ws + off);     off += SHADOW_BYTES_;           // ~17 MB
    uint32_t* bpack   = (uint32_t*)(ws + off);     off += (size_t)8 * NCHUNK_ * 256 * 4;
    float*    partial = (float*)(ws + off);        off += (size_t)B_ * 256 * 4;
    float*    init_change = (float*)(ws + off);    off += 64;
    float*    steps       = (float*)(ws + off);    off += 64;
    int*      converged   = (int*)(ws + off);      off += 64;

    float* out        = (float*)d_out;
    float* out_finals = out;
    float* out_steps  = out + ELEMS_;
    float* out_ponder = out_steps + B_;
    float* out_hist   = out_ponder + 1;

    const int shadow_dwords = (int)((2 * SHADOW_BYTES_) / 4);
    zero_shadows_kernel<<<(shadow_dwords + 255) / 256, 256, 0, stream>>>(
        (uint32_t*)shadowA, shadow_dwords);
    init_state_kernel<<<(ELEMS_ + 255) / 256, 256, 0, stream>>>(
        field, cur, shadowA, init_change, steps, converged, msf);
    pack_weights_kernel<<<(8 * NCHUNK_ * 256 + 255) / 256, 256, 0, stream>>>(
        Wsrc, bpack);

    for (int t = 0; t < max_steps; ++t) {
        const uint32_t* src = (const uint32_t*)((t & 1) ? shadowB : shadowA);
        uint16_t*       dst = (t & 1) ? shadowA : shadowB;
        step_kernel<<<B_ * 256, 256, 0, stream>>>(
            cur, src, dst, bpack, bias, converged, partial, out_finals);
        book_kernel<<<1, 256, 0, stream>>>(
            partial, init_change, steps, converged, out_hist, t);
    }
    finalize_kernel<<<1, 32, 0, stream>>>(steps, out_steps, out_ponder, msf);
}
